// FiLMHeteroConv_13881334301167
// MI455X (gfx1250) — matrix operand order repaired and checked by reading, not compiled, not run on hardware
//
#include <hip/hip_runtime.h>
#include <hip/hip_bf16.h>

#define NODES 100000
#define DIM   128
#define NEDGE 600000
#define NTILES (NODES / 16)   // 6250, exact

typedef __attribute__((ext_vector_type(16))) __bf16 v16bf;
typedef __attribute__((ext_vector_type(8)))  float  v8f;

union Frag16 {
    unsigned short u[16];
    uint4          q[2];
    v16bf          v;
};

__device__ __forceinline__ unsigned short f2bf(float f) {
    unsigned u = __float_as_uint(f);
    u += 0x7FFFu + ((u >> 16) & 1u);          // round-to-nearest-even
    return (unsigned short)(u >> 16);
}

#define LDS_ROW   136                 // 128 + 8 bf16 pad (keeps 16B alignment, spreads banks)
#define LDS_BYTES (384 * LDS_ROW * 2) // 104448 bytes dynamic LDS

// Per-node precompute: for 16-node tiles, D(16x384) = X(16x128) * WC^T via bf16 WMMA.
// WC rows: [0,128) = W -> ylin; [128,256) = GW omega -> film[:,0:128] (sigmoid applied);
//          [256,384) = GW mu  -> film[:,128:256].
__global__ void film_gemm_kernel(const float* __restrict__ x, int batch,
                                 const float* __restrict__ w,
                                 const float* __restrict__ gw,
                                 const float* __restrict__ gb,
                                 unsigned short* __restrict__ ylin,
                                 unsigned short* __restrict__ film) {
    extern __shared__ unsigned short lds[];

    // Cooperative f32->bf16 stage of combined weights (384 x 128) into LDS.
    for (int idx = threadIdx.x; idx < 384 * DIM; idx += blockDim.x) {
        int row = idx >> 7;
        int col = idx & 127;
        float wv = (row < DIM) ? w[row * DIM + col] : gw[(row - DIM) * DIM + col];
        lds[row * LDS_ROW + col] = f2bf(wv);
    }
    __syncthreads();

    const int wave = threadIdx.x >> 5;
    const int lane = threadIdx.x & 31;
    const int g    = lane >> 4;      // lane half (selects K sub-range per ISA layout)
    const int m    = lane & 15;      // A-matrix row within tile
    const int ocol = lane & 15;      // D-matrix column within tile
    const size_t xbase = (size_t)batch * NODES * DIM;

    for (int t = blockIdx.x * 8 + wave; t < NTILES; t += gridDim.x * 8) {
        const int nb = t * 16;
        const float* xr = x + xbase + (size_t)(nb + m) * DIM;

        // A fragments: 16x32 bf16 per K-step. Lane l (row m=l%16, half g=l/16):
        // elems 0..7 -> K = kb + 8g .. +7 ; elems 8..15 -> K = kb + 16 + 8g .. +7
        Frag16 afrag[4];
#pragma unroll
        for (int ks = 0; ks < 4; ++ks) {
            const int kb = 32 * ks;
            float tmp[16];
            *(float4*)&tmp[0]  = *(const float4*)(xr + kb + 8 * g);
            *(float4*)&tmp[4]  = *(const float4*)(xr + kb + 8 * g + 4);
            *(float4*)&tmp[8]  = *(const float4*)(xr + kb + 16 + 8 * g);
            *(float4*)&tmp[12] = *(const float4*)(xr + kb + 16 + 8 * g + 4);
#pragma unroll
            for (int i = 0; i < 16; ++i) afrag[ks].u[i] = f2bf(tmp[i]);
        }

        for (int nt = 0; nt < 24; ++nt) {
            const int o = nt * 16 + ocol;   // global output row of WC held by this lane's column
            // Preload all 4 B fragments first so ds_loads overlap the WMMA chain.
            // B fragment: 32x16 bf16; lane l holds column o, elems 0..7 -> K = 32*ks + 8g + i, elems 8..15 -> K = 32*ks + 16 + 8g + (i-8).
            Frag16 bfrag[4];
            const uint4* bp =
                reinterpret_cast<const uint4*>(&lds[o * LDS_ROW + 8 * g]);
#pragma unroll
            for (int ks = 0; ks < 4; ++ks) {     // +32 elems per ks = +4 uint4
                bfrag[ks].q[0] = bp[4 * ks];
                bfrag[ks].q[1] = bp[4 * ks + 2];
            }
            v8f acc = {0.f, 0.f, 0.f, 0.f, 0.f, 0.f, 0.f, 0.f};
#pragma unroll
            for (int ks = 0; ks < 4; ++ks) {
                acc = __builtin_amdgcn_wmma_f32_16x16x32_bf16(
                    false, afrag[ks].v, false, bfrag[ks].v, (short)0, acc,
                    false, false);
            }
            // D layout: VGPR j, lane l -> (node = nb + j + 8g, column o)
            if (nt < 8) {
#pragma unroll
                for (int j = 0; j < 8; ++j) {
                    int node = nb + j + 8 * g;
                    ylin[(size_t)node * DIM + o] = f2bf(acc[j]);
                }
            } else if (nt < 16) {
                const float bias = gb[o - DIM];
#pragma unroll
                for (int j = 0; j < 8; ++j) {
                    int node = nb + j + 8 * g;
                    float s = acc[j] + bias;
                    s = 1.0f / (1.0f + __expf(-s));
                    film[(size_t)node * (2 * DIM) + (o - DIM)] = f2bf(s);
                }
            } else {
                const float bias = gb[o - DIM];
#pragma unroll
                for (int j = 0; j < 8; ++j) {
                    int node = nb + j + 8 * g;
                    // mu channel (o-256) lands at offset 128+(o-256) == o-128
                    film[(size_t)node * (2 * DIM) + (o - DIM)] = f2bf(acc[j] + bias);
                }
            }
        }
    }
}

// Edge phase: out[dst] += relu(sig_omega[dst] * Ylin[src] + Mu[dst]).
// One wave per edge; 4 channels per thread (uint2 = 4 bf16). Edge indices are
// wave-uniform -> scalar loads.
__global__ void film_edge_kernel(const long long* __restrict__ edges,
                                 const unsigned short* __restrict__ ylin,
                                 const unsigned short* __restrict__ film,
                                 float* __restrict__ out) {
    const int e = blockIdx.x * 8 + threadIdx.y;
    if (e >= NEDGE) return;
    const int c4 = threadIdx.x * 4;               // channel base 0..124
    const int src = (int)edges[e];
    const int dst = (int)edges[NEDGE + e];

    const uint2 yv = *(const uint2*)(ylin + (size_t)src * DIM + c4);
    const uint2 sv = *(const uint2*)(film + (size_t)dst * (2 * DIM) + c4);
    const uint2 mv = *(const uint2*)(film + (size_t)dst * (2 * DIM) + DIM + c4);

    float yl[4], so[4], mu[4];
    yl[0] = __uint_as_float(yv.x << 16); yl[1] = __uint_as_float(yv.x & 0xFFFF0000u);
    yl[2] = __uint_as_float(yv.y << 16); yl[3] = __uint_as_float(yv.y & 0xFFFF0000u);
    so[0] = __uint_as_float(sv.x << 16); so[1] = __uint_as_float(sv.x & 0xFFFF0000u);
    so[2] = __uint_as_float(sv.y << 16); so[3] = __uint_as_float(sv.y & 0xFFFF0000u);
    mu[0] = __uint_as_float(mv.x << 16); mu[1] = __uint_as_float(mv.x & 0xFFFF0000u);
    mu[2] = __uint_as_float(mv.y << 16); mu[3] = __uint_as_float(mv.y & 0xFFFF0000u);

    float* op = out + (size_t)dst * DIM + c4;
#pragma unroll
    for (int i = 0; i < 4; ++i) {
        float v = fmaf(so[i], yl[i], mu[i]);
        v = v > 0.f ? v : 0.f;
        atomicAdd(op + i, v);
    }
}

__global__ void zero_kernel(float4* __restrict__ out, int n4) {
    int i = blockIdx.x * blockDim.x + threadIdx.x;
    if (i < n4) out[i] = make_float4(0.f, 0.f, 0.f, 0.f);
}

extern "C" void kernel_launch(void* const* d_in, const int* in_sizes, int n_in,
                              void* d_out, int out_size, void* d_ws, size_t ws_size,
                              hipStream_t stream) {
    // setup_inputs dict order: x, e0, e1, e2, (w0,gw0,gb0), (w1,gw1,gb1), (w2,gw2,gb2)
    const float* x = (const float*)d_in[0];
    const long long* edges[3] = {(const long long*)d_in[1],
                                 (const long long*)d_in[2],
                                 (const long long*)d_in[3]};
    const float* w[3]  = {(const float*)d_in[4], (const float*)d_in[7], (const float*)d_in[10]};
    const float* gw[3] = {(const float*)d_in[5], (const float*)d_in[8], (const float*)d_in[11]};
    const float* gb[3] = {(const float*)d_in[6], (const float*)d_in[9], (const float*)d_in[12]};

    // Workspace: ylin bf16 [NODES][128] + film bf16 [NODES][256] = 76.8 MB,
    // reused per (relation, batch); fully L2-resident during the edge phase.
    unsigned short* ylin = (unsigned short*)d_ws;
    unsigned short* film = ylin + (size_t)NODES * DIM;

    const int n4 = out_size / 4;  // 6.4M float4
    zero_kernel<<<(n4 + 255) / 256, 256, 0, stream>>>((float4*)d_out, n4);

    for (int r = 0; r < 3; ++r) {
        for (int b = 0; b < 2; ++b) {
            film_gemm_kernel<<<196, 256, LDS_BYTES, stream>>>(
                x, b, w[r], gw[r], gb[r], ylin, film);
            dim3 blk(32, 8);
            film_edge_kernel<<<NEDGE / 8, blk, 0, stream>>>(
                edges[r], ylin, film, (float*)d_out + (size_t)b * NODES * DIM);
        }
    }
}
